// MultiHeadAttention_523986010579
// MI455X (gfx1250) — matrix operand order repaired and checked by reading, not compiled, not run on hardware
//
#include <hip/hip_runtime.h>
#include <hip/hip_bf16.h>

// ---------------------------------------------------------------------------
// MI455X (gfx1250) MultiHeadAttention block.
// GEMMs: v_wmma_f32_16x16x32_bf16, 256 threads / 8 waves, 128x64 C-tile,
// 32x32 per wave. Global->LDS staging uses gfx1250 async-to-LDS copies
// (ASYNCcnt) with double-buffered tiles; f32 lives in LDS, bf16 conversion
// (v_cvt_pk_bf16_f32) happens at fragment build. Global pointers are
// strength-reduced: computed once per thread, advanced by a uniform step
// per K-tile.
// ---------------------------------------------------------------------------

typedef __attribute__((ext_vector_type(16))) __bf16 v16bf;
typedef __attribute__((ext_vector_type(8)))  float  v8f;
typedef __attribute__((ext_vector_type(4)))  int    v4i;

union Frag {
    v16bf          v;
    __bf16         h[16];
    unsigned int   d[8];
};

#define TM 128
#define TN 64
#define TK 32
#define APAD 4   // A row stride 36 f32 = 144B (16B aligned, 16-way bank spread)
#define BPAD 4   // B row stride 36 f32 = 144B
#define ASTR (TK + APAD)
#define BSTR (TK + BPAD)

// ---- gfx1250 async global->LDS copy (ASYNCcnt) with portable fallback -----
// Builtin signature (confirmed via clang diagnostics):
//   __builtin_amdgcn_global_load_async_to_lds_b128(v4i AS1* src, v4i AS3* dst,
//                                                  imm int offset, imm int cpol)
//   __builtin_amdgcn_global_load_async_to_lds_b32 (int AS1* src, int AS3* dst,
//                                                  imm int offset, imm int cpol)
typedef __attribute__((address_space(1))) v4i* g_v4i_ptr;
typedef __attribute__((address_space(3))) v4i* l_v4i_ptr;
typedef __attribute__((address_space(1))) int* g_i32_ptr;
typedef __attribute__((address_space(3))) int* l_i32_ptr;

#if defined(__gfx1250__) && __has_builtin(__builtin_amdgcn_global_load_async_to_lds_b128)
#define HAVE_ASYNC_LDS 1
#endif

__device__ __forceinline__ void cp_b128(const float* g, float* l) {
#if defined(HAVE_ASYNC_LDS)
    // generic LDS address: low 32 bits are the LDS offset (ISA 10.2 aperture map)
    __builtin_amdgcn_global_load_async_to_lds_b128(
        (g_v4i_ptr)(unsigned long long)g,
        (l_v4i_ptr)(unsigned)(unsigned long long)l, 0, 0);
#else
    *(float4*)l = *(const float4*)g;
#endif
}
__device__ __forceinline__ void cp_b32(const float* g, float* l) {
#if defined(HAVE_ASYNC_LDS)
    __builtin_amdgcn_global_load_async_to_lds_b32(
        (g_i32_ptr)(unsigned long long)g,
        (l_i32_ptr)(unsigned)(unsigned long long)l, 0, 0);
#else
    *l = *g;
#endif
}

#if defined(HAVE_ASYNC_LDS)
#if __has_builtin(__builtin_amdgcn_s_wait_asynccnt)
#define ASYNC_WAIT(n) __builtin_amdgcn_s_wait_asynccnt(n)
#else
#define ASYNC_WAIT(n) asm volatile("s_wait_asynccnt %0" :: "i"(n))
#endif
#else
#define ASYNC_WAIT(n) do {} while (0)
#endif

// ---------------------------------------------------------------------------
// Generic batched GEMM:  C = alpha * A @ (B or B^T) + bias + residual
//   batch offset = (bz/bdiv)*S0 + (bz%bdiv)*S1  (expresses b,h head slices)
//   BTRANS (template): B stored (N x K) row-major, contiguous along K.
// All M%128==0, N%64==0, K%32==0 for every call site -> no bounds checks.
// ---------------------------------------------------------------------------
template <int BTRANS>
__global__ __launch_bounds__(256) void gemm_bf16_wmma(
    const float* __restrict__ A, const float* __restrict__ B,
    float* __restrict__ C,
    const float* __restrict__ bias, int bias_per_row,
    const float* __restrict__ residual,
    int M, int N, int K, int lda, int ldb, int ldc,
    int bdiv,
    long long aS0, long long aS1, long long bS0, long long bS1,
    long long cS0, long long cS1,
    float alpha)
{
    __shared__ float Asm_[2][TM][ASTR];   // raw f32 tiles, double buffered
    __shared__ float Bsm_[2][TN][BSTR];   // stored transposed: Bsm_[n][k]

    const int tid  = threadIdx.x;
    const int lane = tid & 31;
    const int wave = tid >> 5;
    const int wm   = wave >> 1;        // 0..3  (M sub-tile)
    const int wn   = wave & 1;         // 0..1  (N sub-tile)

    const int bz = blockIdx.z;
    const long long aoff = (long long)(bz / bdiv) * aS0 + (long long)(bz % bdiv) * aS1;
    const long long boff = (long long)(bz / bdiv) * bS0 + (long long)(bz % bdiv) * bS1;
    const long long coff = (long long)(bz / bdiv) * cS0 + (long long)(bz % bdiv) * cS1;

    const int tile_m = blockIdx.y * TM;
    const int tile_n = blockIdx.x * TN;

    const int ar = tid >> 3;           // 0..31
    const int ac = (tid & 7) << 2;     // 0,4,...,28
    const int bk = tid >> 6;           // 0..3   (non-trans B loader)
    const int bn = tid & 63;           // 0..63

    // ---- loop-invariant strides (SGPR) and per-thread base pointers --------
    const long long lda32 = (long long)lda * 32;
    const long long ldb32 = (long long)ldb * 32;
    const long long ldb4  = (long long)ldb * 4;

    // advanced by a uniform step each K-tile (no per-iteration muls)
    const float* agp = A + aoff + (long long)(tile_m + ar) * lda + ac;
    const float* bgp = BTRANS
        ? B + boff + (long long)(tile_n + ar) * ldb + ac
        : B + boff + (long long)bk * ldb + (tile_n + bn);
    const long long bstep = BTRANS ? (long long)TK : (long long)TK * ldb;

    // async ops issued per tile per wave (compile-time: feeds s_wait_asynccnt)
    constexpr int TILE_OPS = 4 + (BTRANS ? 2 : 8);

    // Issue one (A,B) tile's copies into buffer `bufi`, then advance pointers.
    auto issue_tile = [&](int bufi) {
        float* al = &Asm_[bufi][ar][ac];
        #pragma unroll
        for (int p = 0; p < 4; ++p)                 // A: 128x32, 4 x b128 / thread
            cp_b128(agp + p * lda32, al + p * 32 * ASTR);
        if (BTRANS) {                                // B (NxK): contiguous along K
            float* bl = &Bsm_[bufi][ar][ac];
            #pragma unroll
            for (int p = 0; p < 2; ++p)
                cp_b128(bgp + p * ldb32, bl + p * 32 * BSTR);
        } else {                                     // B (KxN): gather + transpose
            float* bl = &Bsm_[bufi][bn][bk];
            #pragma unroll
            for (int p = 0; p < 8; ++p)
                cp_b32(bgp + p * ldb4, bl + p * 4);
        }
        agp += TK;
        bgp += bstep;
    };

    v8f acc[2][2] = {};

    const int nk = K / TK;
    issue_tile(0);
    int buf = 0;

    for (int kt = 0; kt < nk; ++kt) {
        if (kt + 1 < nk) {
            issue_tile(buf ^ 1);            // stream next tile while computing
            ASYNC_WAIT(TILE_OPS);           // tile kt landed; next still in flight
        } else {
            ASYNC_WAIT(0);
        }
        __syncthreads();                    // cross-wave visibility of buf

        // ---- build bf16 fragments from f32 LDS tile (k = 16*hi + s in A and B) ---
        Frag a0, a1, b0, b1;
        {
            const int row = lane & 15, hi = lane >> 4;
            const float* ap0 = &Asm_[buf][wm * 32 +      row][0];
            const float* ap1 = &Asm_[buf][wm * 32 + 16 + row][0];
            #pragma unroll
            for (int j = 0; j < 8; ++j) {
                const int kp = (hi << 4) + (j << 1);
                const float2 f0 = *(const float2*)(ap0 + kp);
                const float2 f1 = *(const float2*)(ap1 + kp);
                a0.h[2 * j] = (__bf16)f0.x; a0.h[2 * j + 1] = (__bf16)f0.y;
                a1.h[2 * j] = (__bf16)f1.x; a1.h[2 * j + 1] = (__bf16)f1.y;
            }
            const int col = lane & 15, kh = lane >> 4;  // lo lanes K 0-15, hi 16-31
            const float* bp0 = &Bsm_[buf][wn * 32 +      col][kh * 16];
            const float* bp1 = &Bsm_[buf][wn * 32 + 16 + col][kh * 16];
            #pragma unroll
            for (int j = 0; j < 4; ++j) {
                const float4 f0 = *(const float4*)(bp0 + 4 * j);
                const float4 f1 = *(const float4*)(bp1 + 4 * j);
                b0.h[4 * j + 0] = (__bf16)f0.x; b0.h[4 * j + 1] = (__bf16)f0.y;
                b0.h[4 * j + 2] = (__bf16)f0.z; b0.h[4 * j + 3] = (__bf16)f0.w;
                b1.h[4 * j + 0] = (__bf16)f1.x; b1.h[4 * j + 1] = (__bf16)f1.y;
                b1.h[4 * j + 2] = (__bf16)f1.z; b1.h[4 * j + 3] = (__bf16)f1.w;
            }
        }

        acc[0][0] = __builtin_amdgcn_wmma_f32_16x16x32_bf16(false, a0.v, false, b0.v, (short)0, acc[0][0], false, false);
        acc[0][1] = __builtin_amdgcn_wmma_f32_16x16x32_bf16(false, a0.v, false, b1.v, (short)0, acc[0][1], false, false);
        acc[1][0] = __builtin_amdgcn_wmma_f32_16x16x32_bf16(false, a1.v, false, b0.v, (short)0, acc[1][0], false, false);
        acc[1][1] = __builtin_amdgcn_wmma_f32_16x16x32_bf16(false, a1.v, false, b1.v, (short)0, acc[1][1], false, false);

        __syncthreads();                    // all waves done with buf -> reusable
        buf ^= 1;
    }

    // ---- epilogue: alpha, bias, residual, store ----------------------------
    #pragma unroll
    for (int i = 0; i < 2; ++i)
    #pragma unroll
    for (int j = 0; j < 2; ++j) {
        const int col = tile_n + wn * 32 + j * 16 + (lane & 15);
        const int rb  = tile_m + wm * 32 + i * 16 + ((lane >> 4) << 3);
        float*       cp = C + coff + (long long)rb * ldc + col;
        const float* rp = residual ? residual + coff + (long long)rb * ldc + col : nullptr;
        #pragma unroll
        for (int e = 0; e < 8; ++e) {             // VGPR e: M=rb+e
            float val = acc[i][j][e] * alpha;
            if (bias)     val += bias_per_row ? bias[rb + e] : bias[col];
            if (residual) val += rp[(long long)e * ldc];
            cp[(long long)e * ldc] = val;
        }
    }
}

// ---------------------------------------------------------------------------
// Row softmax over Lk=512 with the reference's value-fill masking (1e-9).
// One 256-thread block per (b,h,q) row, 2 elements per thread, in-place.
// ---------------------------------------------------------------------------
__global__ __launch_bounds__(256) void masked_softmax_kernel(
    float* __restrict__ attn, const unsigned char* __restrict__ mask1,
    const unsigned char* __restrict__ mask2, int H, int L)
{
    const int r  = blockIdx.x;           // b*H*L + h*L + q
    const int b  = r / (H * L);
    const int qq = r % L;
    float* row = attn + (long long)r * L;

    const int tid = threadIdx.x, lane = tid & 31, wave = tid >> 5;
    __shared__ float red[8];
    __shared__ float bcast;

    const bool m1 = mask1[b * L + qq] != 0;
    const int i0 = tid, i1 = tid + 256;
    float x0 = row[i0], x1 = row[i1];
    if (m1) { x0 = 1e-9f; x1 = 1e-9f; }
    if (mask2[b * L + i0]) x0 = 1e-9f;
    if (mask2[b * L + i1]) x1 = 1e-9f;

    float m = fmaxf(x0, x1);
    for (int off = 16; off > 0; off >>= 1) m = fmaxf(m, __shfl_xor(m, off, 32));
    if (lane == 0) red[wave] = m;
    __syncthreads();
    if (tid == 0) {
        float t = red[0];
        for (int i = 1; i < 8; ++i) t = fmaxf(t, red[i]);
        bcast = t;
    }
    __syncthreads();
    const float mx = bcast;
    const float e0 = __expf(x0 - mx), e1 = __expf(x1 - mx);
    float s = e0 + e1;
    for (int off = 16; off > 0; off >>= 1) s += __shfl_xor(s, off, 32);
    if (lane == 0) red[wave] = s;
    __syncthreads();
    if (tid == 0) {
        float t = 0.f;
        for (int i = 0; i < 8; ++i) t += red[i];
        bcast = t;
    }
    __syncthreads();
    const float inv = 1.0f / bcast;
    row[i0] = e0 * inv;
    row[i1] = e1 * inv;
}

// ---------------------------------------------------------------------------
// LayerNorm(768) + NaN->0.  One 256-thread block per row, 3 elems/thread.
// ---------------------------------------------------------------------------
__global__ __launch_bounds__(256) void layernorm_kernel(
    const float* __restrict__ x, const float* __restrict__ gamma,
    const float* __restrict__ beta, float* __restrict__ out, int D)
{
    const int r = blockIdx.x;
    const float* px = x + (long long)r * D;
    float* po = out + (long long)r * D;
    const int tid = threadIdx.x, lane = tid & 31, wave = tid >> 5;
    __shared__ float redS[8], redQ[8];
    __shared__ float mu_s, rstd_s;

    const float v0 = px[tid], v1 = px[tid + 256], v2 = px[tid + 512];
    float s = v0 + v1 + v2;
    float q = v0 * v0 + v1 * v1 + v2 * v2;
    for (int off = 16; off > 0; off >>= 1) {
        s += __shfl_xor(s, off, 32);
        q += __shfl_xor(q, off, 32);
    }
    if (lane == 0) { redS[wave] = s; redQ[wave] = q; }
    __syncthreads();
    if (tid == 0) {
        float ts = 0.f, tq = 0.f;
        for (int i = 0; i < 8; ++i) { ts += redS[i]; tq += redQ[i]; }
        const float mu  = ts / (float)D;
        const float var = tq / (float)D - mu * mu;
        mu_s = mu; rstd_s = rsqrtf(var + 1e-6f);
    }
    __syncthreads();
    const float mu = mu_s, rstd = rstd_s;
    const int   idx[3] = { tid, tid + 256, tid + 512 };
    const float vv[3]  = { v0, v1, v2 };
    #pragma unroll
    for (int t = 0; t < 3; ++t) {
        const int c = idx[t];
        const float y = (vv[t] - mu) * rstd * gamma[c] + beta[c];
        po[c] = (y != y) ? 0.0f : y;
    }
}

// ---------------------------------------------------------------------------
extern "C" void kernel_launch(void* const* d_in, const int* in_sizes, int n_in,
                              void* d_out, int out_size, void* d_ws, size_t ws_size,
                              hipStream_t stream)
{
    (void)in_sizes; (void)n_in; (void)out_size; (void)ws_size;

    const float* q     = (const float*)d_in[0];
    const float* k     = (const float*)d_in[1];
    const float* v     = (const float*)d_in[2];
    const float* Wq    = (const float*)d_in[3];
    const float* Wk    = (const float*)d_in[4];
    const float* Wv    = (const float*)d_in[5];
    const float* Wconv = (const float*)d_in[6];
    const float* bconv = (const float*)d_in[7];
    const float* Wfc   = (const float*)d_in[8];
    const float* gamma = (const float*)d_in[9];
    const float* beta  = (const float*)d_in[10];
    const unsigned char* mask1 = (const unsigned char*)d_in[11];  // jnp.bool_ = 1 byte
    const unsigned char* mask2 = (const unsigned char*)d_in[12];

    // B=32, L=512, D_EMB=768, D_K=512, D_V=1024, H=8
    constexpr int Bn = 32, L = 512, DE = 768, DK = 512, DV = 1024, H = 8;
    constexpr long long LL2 = (long long)L * L;             // 262144

    float* ws  = (float*)d_ws;                              // needs ~202 MB
    float* q1  = ws;                                        //  8.39M f32
    float* k1  = ws + (long long)Bn * L * DK;               //  8.39M
    float* v1  = k1 + (long long)Bn * L * DK;               // 16.78M
    float* q1c = v1 + (long long)Bn * L * DV;               //  8.39M
    float* k1c = q1c + (long long)Bn * L * DK;              //  8.39M
    float* ctx = q1;                                        // reuse q1+k1 (dead)
    float* pre = q1c;                                       // reuse q1c+k1c (dead)

    float* outp = (float*)d_out;
    float* attn = outp + (long long)Bn * L * DE;            // second output, (B,H,L,L)

    const dim3 blk(256);
    const float one = 1.0f;
    const float inv_sqrt_dk = 0.04419417382415922f;         // 1/sqrt(512)

    // 1) q1 = q @ Wq   (16384x768 . 768x512)
    gemm_bf16_wmma<0><<<dim3(DK / TN, (Bn * L) / TM, 1), blk, 0, stream>>>(
        q, Wq, q1, nullptr, 0, nullptr,
        Bn * L, DK, DE, DE, DK, DK, 1, 0, 0, 0, 0, 0, 0, one);
    // 2) k1 = k @ Wk
    gemm_bf16_wmma<0><<<dim3(DK / TN, (Bn * L) / TM, 1), blk, 0, stream>>>(
        k, Wk, k1, nullptr, 0, nullptr,
        Bn * L, DK, DE, DE, DK, DK, 1, 0, 0, 0, 0, 0, 0, one);
    // 3) v1 = v @ Wv   (16384x768 . 768x1024)
    gemm_bf16_wmma<0><<<dim3(DV / TN, (Bn * L) / TM, 1), blk, 0, stream>>>(
        v, Wv, v1, nullptr, 0, nullptr,
        Bn * L, DV, DE, DE, DV, DV, 1, 0, 0, 0, 0, 0, 0, one);
    // 4) "conv" over the sequence axis: q1c[b] = Wconv @ q1[b] + bconv[row]
    gemm_bf16_wmma<0><<<dim3(DK / TN, L / TM, Bn), blk, 0, stream>>>(
        Wconv, q1, q1c, bconv, 1, nullptr,
        L, DK, L, L, DK, DK, 1,
        0, 0, (long long)L * DK, 0, (long long)L * DK, 0, one);
    gemm_bf16_wmma<0><<<dim3(DK / TN, L / TM, Bn), blk, 0, stream>>>(
        Wconv, k1, k1c, bconv, 1, nullptr,
        L, DK, L, L, DK, DK, 1,
        0, 0, (long long)L * DK, 0, (long long)L * DK, 0, one);
    // 5) scores[b,h] = (Qh / sqrt(DK)) @ Kh^T   -> attn region of d_out
    gemm_bf16_wmma<1><<<dim3(L / TN, L / TM, Bn * H), blk, 0, stream>>>(
        q1c, k1c, attn, nullptr, 0, nullptr,
        L, L, DK / H, DK, DK, L, H,
        (long long)L * DK, DK / H,            // A: b stride, h stride (col slice)
        (long long)L * DK, DK / H,            // B (transposed use)
        (long long)H * LL2, LL2,              // C: per-(b,h) 512x512
        inv_sqrt_dk);
    // 6) masked softmax over last dim, in place
    masked_softmax_kernel<<<dim3(Bn * H * L), blk, 0, stream>>>(attn, mask1, mask2, H, L);
    // 7) ctx[b, l, h*128+d] = attn[b,h] @ Vh
    gemm_bf16_wmma<0><<<dim3((DV / H) / TN, L / TM, Bn * H), blk, 0, stream>>>(
        attn, v1, ctx, nullptr, 0, nullptr,
        L, DV / H, L, L, DV, DV, H,
        (long long)H * LL2, LL2,
        (long long)L * DV, DV / H,
        (long long)L * DV, DV / H,
        one);
    // 8) pre = ctx @ Wfc + residual(q)   (16384x1024 . 1024x768)
    gemm_bf16_wmma<0><<<dim3(DE / TN, (Bn * L) / TM, 1), blk, 0, stream>>>(
        ctx, Wfc, pre, nullptr, 0, q,
        Bn * L, DE, DV, DV, DE, DE, 1, 0, 0, 0, 0, 0, 0, one);
    // 9) LayerNorm + NaN->0 into the first output
    layernorm_kernel<<<dim3(Bn * L), blk, 0, stream>>>(pre, gamma, beta, outp, DE);
}
